// EncoderDecoder_38302518346285
// MI455X (gfx1250) — matrix operand order repaired and checked by reading, not compiled, not run on hardware
//
#include <hip/hip_runtime.h>

#define HD   512
#define BB   64
#define SS   128
#define VV   32000
#define G4   2048      // 4*H
#define KCAT 1024      // 2*H (x | h)
#define NCHUNK 250
#define CW   128       // vocab cols per chunk

typedef __bf16 bf16_t;
typedef __attribute__((ext_vector_type(8)))  __bf16 v8bf;
typedef __attribute__((ext_vector_type(16))) __bf16 v16bf;
typedef __attribute__((ext_vector_type(8)))  float  v8f;

#if __has_builtin(__builtin_amdgcn_global_load_async_to_lds_b128) && \
    __has_builtin(__builtin_amdgcn_s_wait_asynccnt)
#define USE_ASYNC_LDS 1
typedef int int4v __attribute__((vector_size(16)));
typedef __attribute__((address_space(1))) int4v g_int4;
typedef __attribute__((address_space(3))) int4v l_int4;
#endif

// ---- WMMA fragment loaders (ISA 7.12.2 layouts, wave32) -------------------
// A (16x32 bf16, MxK): lane L holds row M=L&15; lanes 0-15: K = k0+0..7 and
// k0+16..23; lanes 16-31: K = k0+8..15 and k0+24..31.
__device__ __forceinline__ v16bf load_a_frag(const bf16_t* A, int lda, int m0,
                                             int k0, int lane) {
  const int r  = m0 + (lane & 15);
  const int kh = (lane >> 4) << 3;
  const bf16_t* p = A + (size_t)r * lda + (k0 + kh);
  v8bf lo = *reinterpret_cast<const v8bf*>(p);
  v8bf hi = *reinterpret_cast<const v8bf*>(p + 16);
  return __builtin_shufflevector(lo, hi, 0,1,2,3,4,5,6,7,8,9,10,11,12,13,14,15);
}

// B (32x16 bf16, KxN) sourced from row-major weight W[N][K] (B = W^T):
// lane L holds col N=L&15; lanes 0-15: K = k0+0..7 and k0+16..23;
// lanes 16-31: K = k0+8..15 and k0+24..31 (same order as A).
__device__ __forceinline__ v16bf load_b_frag(const bf16_t* W, int ldw, int n0,
                                             int k0, int lane) {
  const int n  = n0 + (lane & 15);
  const int kk = k0 + ((lane >> 4) << 3);
  const bf16_t* p = W + (size_t)n * ldw + kk;
  v8bf lo = *reinterpret_cast<const v8bf*>(p);
  v8bf hi = *reinterpret_cast<const v8bf*>(p + 16);
  return __builtin_shufflevector(lo, hi, 0,1,2,3,4,5,6,7,8,9,10,11,12,13,14,15);
}

__device__ __forceinline__ float sigmoidf_(float x) {
  return 1.0f / (1.0f + __expf(-x));
}

// ---------------------------------------------------------------------------
__global__ void init_kernel(float* __restrict__ h, float* __restrict__ c,
                            float* __restrict__ loss) {
  const int tid = blockIdx.x * blockDim.x + threadIdx.x;
  if (tid == 0) loss[0] = 0.0f;
  for (int i = tid; i < BB * HD; i += gridDim.x * blockDim.x) {
    h[i] = 0.0f; c[i] = 0.0f;
  }
}

// Convert all weights to bf16; LSTM weights as [Wih | Whh] -> Wcat[4H][2H].
__global__ void prep_kernel(const float* __restrict__ eWih, const float* __restrict__ eWhh,
                            const float* __restrict__ dWih, const float* __restrict__ dWhh,
                            const float* __restrict__ aW,   const float* __restrict__ oW,
                            bf16_t* __restrict__ WcatE, bf16_t* __restrict__ WcatD,
                            bf16_t* __restrict__ aWb,   bf16_t* __restrict__ oWb) {
  const long NE = (long)G4 * KCAT;            // 2M
  const long NA = (long)HD * KCAT;            // 0.5M
  const long NO = (long)VV * HD;              // 16.4M
  const long total = NE + NE + NA + NO;
  for (long i = (long)blockIdx.x * blockDim.x + threadIdx.x; i < total;
       i += (long)gridDim.x * blockDim.x) {
    long j = i;
    if (j < NE) {
      const int n = (int)(j >> 10), k = (int)(j & (KCAT - 1));
      const float v = (k < HD) ? eWih[(size_t)n * HD + k]
                               : eWhh[(size_t)n * HD + (k - HD)];
      WcatE[j] = (bf16_t)v;
    } else if ((j -= NE) < NE) {
      const int n = (int)(j >> 10), k = (int)(j & (KCAT - 1));
      const float v = (k < HD) ? dWih[(size_t)n * HD + k]
                               : dWhh[(size_t)n * HD + (k - HD)];
      WcatD[j] = (bf16_t)v;
    } else if ((j -= NE) < NA) {
      aWb[j] = (bf16_t)aW[j];
    } else {
      j -= NA;
      oWb[j] = (bf16_t)oW[j];
    }
  }
}

// ---- Encoder: one WGP, 32 waves, loops all 128 timesteps ------------------
// Dynamic LDS: catA = [x | h] bf16 [64][1024]  (128 KB)
__global__ __launch_bounds__(1024) void enc_kernel(
    const int* __restrict__ source, const float* __restrict__ embed_src,
    const bf16_t* __restrict__ Wcat, const float* __restrict__ bih,
    const float* __restrict__ bhh, float* __restrict__ h, float* __restrict__ c,
    float* __restrict__ gates, float* __restrict__ enc_hs) {
  extern __shared__ char smem[];
  bf16_t* catA = (bf16_t*)smem;
  const int tid = threadIdx.x;
  const int lane = tid & 31, wave = tid >> 5;
  for (int s = 0; s < SS; ++s) {
    // stage 1: A = [embed(x) | h] in bf16 (LDS)
    for (int idx = tid; idx < BB * KCAT; idx += 1024) {
      const int b = idx >> 10, k = idx & (KCAT - 1);
      float v;
      if (k < HD) { const int w = source[b * SS + s]; v = embed_src[(size_t)w * HD + k]; }
      else          v = h[(b << 9) + (k - HD)];
      catA[idx] = (bf16_t)v;
    }
    __syncthreads();
    // stage 2: gates = A @ Wcat^T + bias  (512 tiles, 16 per wave)
    for (int ti = 0; ti < 16; ++ti) {
      const int t = wave * 16 + ti;
      const int rt = t >> 7, ct = t & 127;
      v8f acc = {};
      #pragma unroll 4
      for (int kk = 0; kk < KCAT; kk += 32) {
        v16bf a  = load_a_frag(catA, KCAT, rt * 16, kk, lane);
        v16bf bm = load_b_frag(Wcat, KCAT, ct * 16, kk, lane);
        acc = __builtin_amdgcn_wmma_f32_16x16x32_bf16(false, a, false, bm,
                                                      (short)0, acc, false, false);
      }
      const int n  = ct * 16 + (lane & 15);
      const int m0 = rt * 16 + ((lane >> 4) << 3);
      const float bias = bih[n] + bhh[n];
      #pragma unroll
      for (int r = 0; r < 8; ++r)
        gates[(size_t)(m0 + r) * G4 + n] = acc[r] + bias;
    }
    __threadfence_block(); __syncthreads();
    // stage 3: LSTM elementwise + padding mask
    for (int idx = tid; idx < BB * HD; idx += 1024) {
      const int b = idx >> 9, j = idx & (HD - 1);
      const int w = source[b * SS + s];
      const float gi = gates[(size_t)b * G4 + j];
      const float gf = gates[(size_t)b * G4 + HD + j];
      const float gg = gates[(size_t)b * G4 + 2 * HD + j];
      const float go = gates[(size_t)b * G4 + 3 * HD + j];
      const float c0 = c[idx], h0 = h[idx];
      const float c2 = sigmoidf_(gf) * c0 + sigmoidf_(gi) * tanhf(gg);
      const float h2 = sigmoidf_(go) * tanhf(c2);
      const float hn = (w != 0) ? h2 : h0;
      const float cn = (w != 0) ? c2 : c0;
      h[idx] = hn; c[idx] = cn;
      enc_hs[((size_t)s * BB + b) * HD + j] = hn;
    }
    __threadfence_block(); __syncthreads();
  }
}

// ---- Decoder step: LSTM + attention + ha projection (one WGP) -------------
// Dynamic LDS: catA bf16 [64][1024] (128 KB) + sc f32 [128][64] (32 KB)
__global__ __launch_bounds__(1024) void dec_step_kernel(
    int step, const int* __restrict__ source, const int* __restrict__ target,
    const float* __restrict__ embed_tgt, const bf16_t* __restrict__ Wcat,
    const float* __restrict__ bih, const float* __restrict__ bhh,
    float* __restrict__ h, float* __restrict__ c,
    float* __restrict__ gates, const float* __restrict__ enc_hs,
    const bf16_t* __restrict__ aWb, const float* __restrict__ attn_b,
    bf16_t* __restrict__ haB) {
  extern __shared__ char smem[];
  bf16_t* catA = (bf16_t*)smem;
  float*  sc   = (float*)(smem + (size_t)BB * KCAT * 2);
  const int tid = threadIdx.x, lane = tid & 31, wave = tid >> 5;
  // stage 1: A = [embed(y) | h]
  for (int idx = tid; idx < BB * KCAT; idx += 1024) {
    const int b = idx >> 10, k = idx & (KCAT - 1);
    float v;
    if (k < HD) { const int w = target[b * SS + step]; v = embed_tgt[(size_t)w * HD + k]; }
    else          v = h[(b << 9) + (k - HD)];
    catA[idx] = (bf16_t)v;
  }
  __syncthreads();
  // stage 2: gate GEMM
  for (int ti = 0; ti < 16; ++ti) {
    const int t = wave * 16 + ti;
    const int rt = t >> 7, ct = t & 127;
    v8f acc = {};
    #pragma unroll 4
    for (int kk = 0; kk < KCAT; kk += 32) {
      v16bf a  = load_a_frag(catA, KCAT, rt * 16, kk, lane);
      v16bf bm = load_b_frag(Wcat, KCAT, ct * 16, kk, lane);
      acc = __builtin_amdgcn_wmma_f32_16x16x32_bf16(false, a, false, bm,
                                                    (short)0, acc, false, false);
    }
    const int n  = ct * 16 + (lane & 15);
    const int m0 = rt * 16 + ((lane >> 4) << 3);
    const float bias = bih[n] + bhh[n];
    #pragma unroll
    for (int r = 0; r < 8; ++r)
      gates[(size_t)(m0 + r) * G4 + n] = acc[r] + bias;
  }
  __threadfence_block(); __syncthreads();
  // stage 3: LSTM elementwise + mask; stash h2 (bf16) for the ha GEMM
  for (int idx = tid; idx < BB * HD; idx += 1024) {
    const int b = idx >> 9, j = idx & (HD - 1);
    const int w = target[b * SS + step];
    const float gi = gates[(size_t)b * G4 + j];
    const float gf = gates[(size_t)b * G4 + HD + j];
    const float gg = gates[(size_t)b * G4 + 2 * HD + j];
    const float go = gates[(size_t)b * G4 + 3 * HD + j];
    const float c0 = c[idx], h0 = h[idx];
    const float c2 = sigmoidf_(gf) * c0 + sigmoidf_(gi) * tanhf(gg);
    const float h2 = sigmoidf_(go) * tanhf(c2);
    const float hn = (w != 0) ? h2 : h0;
    const float cn = (w != 0) ? c2 : c0;
    h[idx] = hn; c[idx] = cn;
    catA[(b << 10) + HD + j] = (bf16_t)hn;
  }
  __threadfence_block(); __syncthreads();
  // stage 4: attention scores (batched dot, masked by source padding)
  for (int idx = tid; idx < SS * BB; idx += 1024) {
    const int st = idx >> 6, b = idx & 63;
    const float* hp = h + (b << 9);
    const float* ep = enc_hs + ((size_t)st * BB + b) * HD;
    float acc = 0.0f;
    for (int k = 0; k < HD; k += 4)
      acc += hp[k] * ep[k] + hp[k + 1] * ep[k + 1] +
             hp[k + 2] * ep[k + 2] + hp[k + 3] * ep[k + 3];
    sc[st * BB + b] = (source[b * SS + st] != 0) ? acc : -__builtin_inff();
  }
  __syncthreads();
  // stage 5: softmax over source positions
  if (tid < BB) {
    const int b = tid;
    float m = -__builtin_inff();
    for (int st = 0; st < SS; ++st) m = fmaxf(m, sc[st * BB + b]);
    float sum = 0.0f;
    for (int st = 0; st < SS; ++st) {
      const float x = sc[st * BB + b];
      if (x != -__builtin_inff()) sum += __expf(x - m);
    }
    const float inv = (sum > 0.0f) ? 1.0f / sum : 0.0f;
    for (int st = 0; st < SS; ++st) {
      const float x = sc[st * BB + b];
      sc[st * BB + b] = (x != -__builtin_inff()) ? __expf(x - m) * inv : 0.0f;
    }
  }
  __syncthreads();
  // stage 6: context vector -> catA[:, 0:H]
  for (int idx = tid; idx < BB * HD; idx += 1024) {
    const int b = idx >> 9, j = idx & (HD - 1);
    float acc = 0.0f;
    for (int st = 0; st < SS; ++st)
      acc += sc[st * BB + b] * enc_hs[((size_t)st * BB + b) * HD + j];
    catA[(b << 10) + j] = (bf16_t)acc;
  }
  __syncthreads();
  // stage 7: ha = tanh([ctx | h2] @ attn_W^T + b)  (128 tiles, 4 per wave)
  for (int ti = 0; ti < 4; ++ti) {
    const int t = wave * 4 + ti;
    const int rt = t >> 5, ct = t & 31;
    v8f acc = {};
    #pragma unroll 4
    for (int kk = 0; kk < KCAT; kk += 32) {
      v16bf a  = load_a_frag(catA, KCAT, rt * 16, kk, lane);
      v16bf bm = load_b_frag(aWb, KCAT, ct * 16, kk, lane);
      acc = __builtin_amdgcn_wmma_f32_16x16x32_bf16(false, a, false, bm,
                                                    (short)0, acc, false, false);
    }
    const int n  = ct * 16 + (lane & 15);
    const int m0 = rt * 16 + ((lane >> 4) << 3);
    const float bias = attn_b[n];
    #pragma unroll
    for (int r = 0; r < 8; ++r)
      haB[(size_t)(m0 + r) * HD + n] = (bf16_t)tanhf(acc[r] + bias);
  }
}

// ---- Vocab projection chunk + online log-softmax partials -----------------
// Dynamic LDS: aS bf16 [64][512] (64 KB, async-staged) + lg f32 [64][128] (32 KB)
__global__ __launch_bounds__(256) void logits_kernel(
    int step, const int* __restrict__ target, const bf16_t* __restrict__ haB,
    const bf16_t* __restrict__ oWb, const float* __restrict__ out_b,
    float* __restrict__ part_m, float* __restrict__ part_s,
    float* __restrict__ lt) {
  extern __shared__ char smem[];
  bf16_t* aS = (bf16_t*)smem;
  float*  lg = (float*)(smem + (size_t)BB * HD * 2);
  const int tid = threadIdx.x, lane = tid & 31, wave = tid >> 5;
  const int cbase = blockIdx.x * CW;
  // stage A (ha matrix, 64 KB) into LDS once per workgroup
#if defined(USE_ASYNC_LDS)
  for (int i = tid; i < (BB * HD) / 8; i += 256) {   // 16B chunks
    const bf16_t* g = haB + (size_t)i * 8;
    bf16_t*       l = aS + (size_t)i * 8;
    __builtin_amdgcn_global_load_async_to_lds_b128(
        (g_int4*)(size_t)(const void*)g,
        (l_int4*)(unsigned)(size_t)(void*)l, 0, 0);
  }
  __builtin_amdgcn_s_wait_asynccnt(0);
  __syncthreads();
#else
  for (int i = tid; i < (BB * HD) / 8; i += 256)
    reinterpret_cast<v8bf*>(aS)[i] = reinterpret_cast<const v8bf*>(haB)[i];
  __syncthreads();
#endif
  for (int ti = 0; ti < 4; ++ti) {
    const int t = wave * 4 + ti;                     // 32 tiles: 4 rows x 8 cols
    const int rt = t >> 3, ctl = t & 7;
    const int n0 = cbase + ctl * 16;
    __builtin_prefetch((const void*)(oWb + (size_t)(n0 + (lane & 15)) * HD), 0, 3);
    v8f acc = {};
    #pragma unroll 4
    for (int kk = 0; kk < HD; kk += 32) {
      v16bf a  = load_a_frag(aS, HD, rt * 16, kk, lane);
      v16bf bm = load_b_frag(oWb, HD, n0, kk, lane);
      acc = __builtin_amdgcn_wmma_f32_16x16x32_bf16(false, a, false, bm,
                                                    (short)0, acc, false, false);
    }
    const int nl = ctl * 16 + (lane & 15);
    const int m0 = rt * 16 + ((lane >> 4) << 3);
    const float bias = out_b[cbase + nl];
    #pragma unroll
    for (int r = 0; r < 8; ++r) lg[(m0 + r) * CW + nl] = acc[r] + bias;
  }
  __syncthreads();
  if (tid < BB) {
    const int b = tid;
    float m = -__builtin_inff();
    for (int j = 0; j < CW; ++j) m = fmaxf(m, lg[b * CW + j]);
    float sum = 0.0f;
    for (int j = 0; j < CW; ++j) sum += __expf(lg[b * CW + j] - m);
    part_m[blockIdx.x * BB + b] = m;
    part_s[blockIdx.x * BB + b] = sum;
    const int tgt = target[b * SS + step + 1];
    if (tgt >= cbase && tgt < cbase + CW) lt[b] = lg[b * CW + (tgt - cbase)];
  }
}

// ---- Merge partials, accumulate masked mean NLL ---------------------------
__global__ __launch_bounds__(64) void loss_kernel(
    int step, const int* __restrict__ target, const float* __restrict__ part_m,
    const float* __restrict__ part_s, const float* __restrict__ lt,
    float* __restrict__ loss) {
  __shared__ float stok[BB];
  __shared__ int sval[BB];
  const int b = threadIdx.x;
  if (b < BB) {
    float m = -__builtin_inff();
    for (int ch = 0; ch < NCHUNK; ++ch) m = fmaxf(m, part_m[ch * BB + b]);
    float sum = 0.0f;
    for (int ch = 0; ch < NCHUNK; ++ch)
      sum += part_s[ch * BB + b] * __expf(part_m[ch * BB + b] - m);
    const int tgt = target[b * SS + step + 1];
    const float lp = lt[b] - m - __logf(sum);
    stok[b] = (tgt != 0) ? -lp : 0.0f;
    sval[b] = (tgt != 0) ? 1 : 0;
  }
  __syncthreads();
  if (b == 0) {
    float s = 0.0f; int cnt = 0;
    for (int i = 0; i < BB; ++i) { s += stok[i]; cnt += sval[i]; }
    loss[0] += s / fmaxf((float)cnt, 1.0f);
  }
}

// ---------------------------------------------------------------------------
extern "C" void kernel_launch(void* const* d_in, const int* in_sizes, int n_in,
                              void* d_out, int out_size, void* d_ws,
                              size_t ws_size, hipStream_t stream) {
  (void)in_sizes; (void)n_in; (void)out_size; (void)ws_size;
  const int*   source    = (const int*)  d_in[0];
  const int*   target    = (const int*)  d_in[1];
  const float* embed_src = (const float*)d_in[2];
  const float* embed_tgt = (const float*)d_in[3];
  const float* enc_Wih   = (const float*)d_in[4];
  const float* enc_Whh   = (const float*)d_in[5];
  const float* enc_bih   = (const float*)d_in[6];
  const float* enc_bhh   = (const float*)d_in[7];
  const float* dec_Wih   = (const float*)d_in[8];
  const float* dec_Whh   = (const float*)d_in[9];
  const float* dec_bih   = (const float*)d_in[10];
  const float* dec_bhh   = (const float*)d_in[11];
  const float* attn_W    = (const float*)d_in[12];
  const float* attn_b    = (const float*)d_in[13];
  const float* out_W     = (const float*)d_in[14];
  const float* out_b     = (const float*)d_in[15];
  float* loss = (float*)d_out;

  char* base = (char*)d_ws;
  size_t off = 0;
  auto alloc = [&](size_t bytes) -> char* {
    char* p = base + off;
    off = (off + bytes + 255) & ~(size_t)255;
    return p;
  };
  bf16_t* WcatE  = (bf16_t*)alloc((size_t)G4 * KCAT * 2);
  bf16_t* WcatD  = (bf16_t*)alloc((size_t)G4 * KCAT * 2);
  bf16_t* aWb    = (bf16_t*)alloc((size_t)HD * KCAT * 2);
  bf16_t* oWb    = (bf16_t*)alloc((size_t)VV * HD * 2);
  float*  enc_hs = (float*) alloc((size_t)SS * BB * HD * 4);
  float*  hbuf   = (float*) alloc((size_t)BB * HD * 4);
  float*  cbuf   = (float*) alloc((size_t)BB * HD * 4);
  float*  gates  = (float*) alloc((size_t)BB * G4 * 4);
  bf16_t* haB    = (bf16_t*)alloc((size_t)BB * HD * 2);
  float*  part_m = (float*) alloc((size_t)NCHUNK * BB * 4);
  float*  part_s = (float*) alloc((size_t)NCHUNK * BB * 4);
  float*  lt     = (float*) alloc((size_t)BB * 4);

  const size_t lds_enc    = (size_t)BB * KCAT * 2;                       // 128 KB
  const size_t lds_dec    = (size_t)BB * KCAT * 2 + (size_t)SS * BB * 4; // 160 KB
  const size_t lds_logits = (size_t)BB * HD * 2 + (size_t)BB * CW * 4;   // 96 KB

  init_kernel<<<64, 256, 0, stream>>>(hbuf, cbuf, loss);
  prep_kernel<<<4096, 256, 0, stream>>>(enc_Wih, enc_Whh, dec_Wih, dec_Whh,
                                        attn_W, out_W, WcatE, WcatD, aWb, oWb);
  enc_kernel<<<1, 1024, lds_enc, stream>>>(source, embed_src, WcatE, enc_bih,
                                           enc_bhh, hbuf, cbuf, gates, enc_hs);
  for (int t = 0; t < SS - 1; ++t) {
    dec_step_kernel<<<1, 1024, lds_dec, stream>>>(t, source, target, embed_tgt,
                                                  WcatD, dec_bih, dec_bhh, hbuf,
                                                  cbuf, gates, enc_hs, aWb,
                                                  attn_b, haB);
    logits_kernel<<<NCHUNK, 256, lds_logits, stream>>>(t, target, haB, oWb,
                                                       out_b, part_m, part_s, lt);
    loss_kernel<<<1, 64, 0, stream>>>(t, target, part_m, part_s, lt, loss);
  }
}
